// MultiHeadAttention_283467841566
// MI455X (gfx1250) — matrix operand order repaired and checked by reading, not compiled, not run on hardware
//
#include <hip/hip_runtime.h>

typedef _Float16 f16;
typedef __attribute__((ext_vector_type(16))) _Float16 v16h;
typedef __attribute__((ext_vector_type(8)))  _Float16 v8h;
typedef __attribute__((ext_vector_type(8)))  float    v8f;
typedef __attribute__((ext_vector_type(4)))  float    v4f;

#define BATCH 4
#define S_LEN 2048
#define DM    1024
#define NH    16
#define DEPTH 64

__device__ __forceinline__ v8f wmma_f16(v16h a, v16h b, v8f c) {
  return __builtin_amdgcn_wmma_f32_16x16x32_f16(false, a, false, b, (short)0, c, false, false);
}

// Build A fragment (16x32 f16): lane-lo rows get K {0..7,16..23}, lane-hi {8..15,24..31}
__device__ __forceinline__ v16h load_fragA_f16(const f16* __restrict__ rowp, int lhi) {
  const int o = lhi ? 8 : 0;
  v8h x0 = *(const v8h*)(rowp + o);
  v8h x1 = *(const v8h*)(rowp + o + 16);
  v16h r;
#pragma unroll
  for (int e = 0; e < 8; ++e) { r[e] = x0[e]; r[e + 8] = x1[e]; }
  return r;
}

// Build B fragment (32x16 f16): lane-lo cols get K {0..7,16..23}, lane-hi {8..15,24..31}
__device__ __forceinline__ v16h load_fragB_f16(const f16* __restrict__ colp_k0, int lhi) {
  const f16* p = colp_k0 + (lhi ? 8 : 0);
  v8h x0 = *(const v8h*)(p);
  v8h x1 = *(const v8h*)(p + 16);
  v16h r;
#pragma unroll
  for (int e = 0; e < 8; ++e) { r[e] = x0[e]; r[e + 8] = x1[e]; }
  return r;
}

// ---------------------------------------------------------------------------
// Unified WMMA GEMM, M=8192 N=1024 K=1024, block tile 128x128x64, 8 waves.
// MODE 0: C = A32 @ W + bias, f16 out in [B,H,S,64]   (Q / K projection)
// MODE 1: C = A32 @ W + bias, f16 out in [B,H,64,S]   (V projection, transposed)
// MODE 2: C = A16(ctx,[B,H,S,64]) @ W + bias, f32 out [M,N]  (output projection)
// ---------------------------------------------------------------------------
template <int MODE>
__global__ void __launch_bounds__(256)
gemm_kernel(const float* __restrict__ A32, const f16* __restrict__ A16,
            const float* __restrict__ W, const float* __restrict__ bias,
            f16* __restrict__ out16, float* __restrict__ out32) {
  constexpr int BM = 128, BN = 128, BK = 64;
  constexpr int LSTR = 72;  // f16 row stride (conflict-spreading, 16B aligned)
  __shared__ f16 lA[BM * LSTR];  // [m][k]
  __shared__ f16 lB[BN * LSTR];  // transposed: [n][k]

  const int tid  = threadIdx.x;
  const int bm   = blockIdx.x * BM;
  const int bn   = blockIdx.y * BN;
  const int w    = tid >> 5;
  const int lane = tid & 31;
  const int wm   = (w & 3) * 32;   // 4 waves along M
  const int wn   = (w >> 2) * 64;  // 2 waves along N
  const int lhi  = (lane >> 4) & 1;
  const int l15  = lane & 15;

  v8f acc[2][4];
#pragma unroll
  for (int i = 0; i < 2; ++i)
#pragma unroll
    for (int j = 0; j < 4; ++j) acc[i][j] = (v8f)0.0f;

  for (int k0 = 0; k0 < DM; k0 += BK) {
    __syncthreads();
    // ---- stage A tile (128 x 64) into LDS as f16
    if (MODE == 2) {
      // ctx f16, head-major; BK==DEPTH so the whole k-tile lies in head h=k0/64
      const int h = k0 >> 6;
#pragma unroll
      for (int i = 0; i < 4; ++i) {
        int idx = tid + i * 256;        // 1024 segments of 8 f16
        int r = idx >> 3, sc = idx & 7; // row, 8-elem segment
        int m = bm + r;
        int bb = m >> 11, ss = m & (S_LEN - 1);
        v8h vv = *(const v8h*)(A16 + ((size_t)(bb * NH + h) * S_LEN + ss) * DEPTH + sc * 8);
        *(v8h*)(lA + r * LSTR + sc * 8) = vv;
      }
    } else {
#pragma unroll
      for (int i = 0; i < 8; ++i) {
        int idx = tid + i * 256;         // 2048 float4
        int r = idx >> 4, c4 = idx & 15;
        v4f v = *(const v4f*)(A32 + (size_t)(bm + r) * DM + k0 + c4 * 4);
        f16* dst = lA + r * LSTR + c4 * 4;
        dst[0] = (f16)v.x; dst[1] = (f16)v.y; dst[2] = (f16)v.z; dst[3] = (f16)v.w;
      }
    }
    // ---- stage B tile (64 x 128) transposed into LDS as f16 [n][k]
#pragma unroll
    for (int i = 0; i < 8; ++i) {
      int idx = tid + i * 256;
      int kr = idx >> 5, c4 = idx & 31;
      v4f v = *(const v4f*)(W + (size_t)(k0 + kr) * DM + bn + c4 * 4);
      lB[(c4 * 4 + 0) * LSTR + kr] = (f16)v.x;
      lB[(c4 * 4 + 1) * LSTR + kr] = (f16)v.y;
      lB[(c4 * 4 + 2) * LSTR + kr] = (f16)v.z;
      lB[(c4 * 4 + 3) * LSTR + kr] = (f16)v.w;
    }
    __syncthreads();
    // ---- compute: 2 k-chunks of 32
#pragma unroll
    for (int c = 0; c < 2; ++c) {
      v16h af[2], bf[4];
#pragma unroll
      for (int mt = 0; mt < 2; ++mt)
        af[mt] = load_fragA_f16(lA + (wm + mt * 16 + l15) * LSTR + c * 32, lhi);
#pragma unroll
      for (int nt = 0; nt < 4; ++nt)
        bf[nt] = load_fragB_f16(lB + (wn + nt * 16 + l15) * LSTR + c * 32, lhi);
#pragma unroll
      for (int mt = 0; mt < 2; ++mt)
#pragma unroll
        for (int nt = 0; nt < 4; ++nt)
          acc[mt][nt] = wmma_f16(af[mt], bf[nt], acc[mt][nt]);
    }
  }

  // ---- epilogue
#pragma unroll
  for (int mt = 0; mt < 2; ++mt)
#pragma unroll
    for (int nt = 0; nt < 4; ++nt) {
      int n = bn + wn + nt * 16 + l15;
      float bvl = bias[n];
#pragma unroll
      for (int r = 0; r < 8; ++r) {
        int m = bm + wm + mt * 16 + (lhi ? 8 + r : r);
        float val = acc[mt][nt][r] + bvl;
        int bb = m >> 11, ss = m & (S_LEN - 1);
        int hh = n >> 6, dd = n & (DEPTH - 1);
        if (MODE == 0)
          out16[((size_t)(bb * NH + hh) * S_LEN + ss) * DEPTH + dd] = (f16)val;
        else if (MODE == 1)
          out16[((size_t)(bb * NH + hh) * DEPTH + dd) * S_LEN + ss] = (f16)val;
        else
          out32[(size_t)m * DM + n] = val;
      }
    }
}

// ---------------------------------------------------------------------------
// Fused attention: one block = (b, h, 16-query strip).
// 8 waves, each owns a 256-key segment (segment id made SCALAR via
// readfirstlane so causal skipping is an s_cbranch, never an EXEC change
// around WMMA). logits(16 x 2048, f32) live in LDS:
// QK^T (WMMA) -> scale+causal mask -> softmax in LDS -> attn written to HBM
// once -> attn@V (WMMA, probs read back from LDS) -> ds_add_f32 reduction.
// ---------------------------------------------------------------------------
__global__ void __launch_bounds__(256)
attn_kernel(const f16* __restrict__ Qh, const f16* __restrict__ Kh,
            const f16* __restrict__ Vt, float* __restrict__ attn,
            f16* __restrict__ ctx) {
  constexpr int LROW = 2064;  // padded f32 row stride
  extern __shared__ float sm[];
  float* logits = sm;                 // 16 * LROW
  float* cbuf   = sm + 16 * LROW;     // 16 * DEPTH ctx accumulator

  const int tid  = threadIdx.x;
  const int lane = tid & 31;
  const int lhi  = lane >> 4;
  const int l15  = lane & 15;
  const int qb   = blockIdx.x * 16;
  const int h    = blockIdx.y;
  const int b    = blockIdx.z;
  const size_t bh = (size_t)b * NH + h;
  const f16* Qp = Qh + (bh * S_LEN + qb) * DEPTH;
  const f16* Kp = Kh + bh * S_LEN * DEPTH;
  const f16* Vp = Vt + bh * DEPTH * S_LEN;
  const float scale = 0.125f;  // 1/sqrt(64)

  // scalar (SGPR) wave id -> scalar causal bounds, EXEC untouched around WMMA
  const int wsc  = __builtin_amdgcn_readfirstlane(tid) >> 5;
  const int seg  = wsc * 256;
  const int qmax = qb + 15;

  for (int i = tid; i < 16 * DEPTH; i += 256) cbuf[i] = 0.0f;

  // ---- cooperative fill of the fully-masked column range [qb+16, 2048)
  {
    const int c0 = qb + 16;
    const v4f neg = {-1e9f, -1e9f, -1e9f, -1e9f};
    for (int rrow = 0; rrow < 16; ++rrow) {
      float* lr = logits + rrow * LROW;
      for (int cc = c0 + tid * 4; cc < S_LEN; cc += 1024)
        *(v4f*)(lr + cc) = neg;
    }
  }

  // Q fragments for this strip (row = l15), 2 depth chunks
  v16h aq[2];
#pragma unroll
  for (int c = 0; c < 2; ++c)
    aq[c] = load_fragA_f16(Qp + l15 * DEPTH + c * 32, lhi);

  // ---- logits = Q K^T * scale + causal mask (live tiles only, scalar bound)
  int ntiles = 0;
  if (seg <= qmax) {
    ntiles = ((qmax - seg) >> 4) + 1;
    if (ntiles > 16) ntiles = 16;
  }
  for (int nt = 0; nt < ntiles; ++nt) {
    const int kb = seg + nt * 16;           // scalar
    v8f acc = (v8f)0.0f;
#pragma unroll
    for (int c = 0; c < 2; ++c) {
      v16h bfr = load_fragB_f16(Kp + (size_t)(kb + l15) * DEPTH + c * 32, lhi);
      acc = wmma_f16(aq[c], bfr, acc);
    }
    if (kb + 15 > qb) {  // tile straddles the diagonal: apply per-element mask
#pragma unroll
      for (int r = 0; r < 8; ++r) {
        int m = lhi ? 8 + r : r;
        int kpos = kb + l15;
        float v = acc[r] * scale;
        if (kpos > qb + m) v -= 1e9f;
        logits[m * LROW + kpos] = v;
      }
    } else {             // fully unmasked tile: plain scaled store
#pragma unroll
      for (int r = 0; r < 8; ++r) {
        int m = lhi ? 8 + r : r;
        logits[m * LROW + kb + l15] = acc[r] * scale;
      }
    }
  }
  __syncthreads();

  // ---- softmax: 16 threads per row (within one 16-lane shuffle group)
  {
    const int row = tid >> 4, j = tid & 15;
    float* lr = logits + row * LROW;
    float mx = -1e30f;
    for (int i = 0; i < 128; ++i) mx = fmaxf(mx, lr[j + i * 16]);
#pragma unroll
    for (int d = 8; d > 0; d >>= 1) mx = fmaxf(mx, __shfl_xor(mx, d, 16));
    float s = 0.0f;
    for (int i = 0; i < 128; ++i) s += __expf(lr[j + i * 16] - mx);
#pragma unroll
    for (int d = 8; d > 0; d >>= 1) s += __shfl_xor(s, d, 16);
    float inv = 1.0f / s;
    float* ar = attn + (bh * S_LEN + qb + row) * S_LEN;
    for (int i = 0; i < 128; ++i) {
      int col = j + i * 16;
      float p = __expf(lr[col] - mx) * inv;
      lr[col] = p;      // probs kept in LDS for attn@V
      ar[col] = p;      // attn output: written to HBM exactly once
    }
  }
  __syncthreads();

  // ---- ctx = attn @ V over this wave's key segment (scalar chunk bound)
  v8f accd[4];
#pragma unroll
  for (int dt = 0; dt < 4; ++dt) accd[dt] = (v8f)0.0f;
  int nch = 0;
  if (seg <= qmax) {
    nch = ((qmax - seg) >> 5) + 1;
    if (nch > 8) nch = 8;
  }
  for (int kc = 0; kc < nch; ++kc) {
    const int kb = seg + kc * 32;           // scalar
    v16h af;
    {
      const float* pr = logits + l15 * LROW + kb;
      const int o = lhi ? 8 : 0;
#pragma unroll
      for (int e = 0; e < 8; ++e) af[e]     = (f16)pr[o + e];
#pragma unroll
      for (int e = 0; e < 8; ++e) af[e + 8] = (f16)pr[o + 16 + e];
    }
#pragma unroll
    for (int dt = 0; dt < 4; ++dt) {
      v16h bfr = load_fragB_f16(Vp + (size_t)(dt * 16 + l15) * S_LEN + kb, lhi);
      accd[dt] = wmma_f16(af, bfr, accd[dt]);
    }
  }
#pragma unroll
  for (int dt = 0; dt < 4; ++dt)
#pragma unroll
    for (int r = 0; r < 8; ++r) {
      int m = lhi ? 8 + r : r;
      atomicAdd(&cbuf[m * DEPTH + dt * 16 + l15], accd[dt][r]);  // ds_add_f32
    }
  __syncthreads();

  f16* cp = ctx + (bh * S_LEN + qb) * DEPTH;
  for (int i = tid; i < 16 * DEPTH; i += 256) cp[i] = (f16)cbuf[i];
}

// ---------------------------------------------------------------------------
extern "C" void kernel_launch(void* const* d_in, const int* in_sizes, int n_in,
                              void* d_out, int out_size, void* d_ws, size_t ws_size,
                              hipStream_t stream) {
  (void)in_sizes; (void)n_in; (void)out_size; (void)ws_size;
  const float* q  = (const float*)d_in[0];
  const float* k  = (const float*)d_in[1];
  const float* v  = (const float*)d_in[2];
  // d_in[3] = mask (causal) — applied analytically in attn_kernel
  const float* wq = (const float*)d_in[4];
  const float* bq = (const float*)d_in[5];
  const float* wk = (const float*)d_in[6];
  const float* bk = (const float*)d_in[7];
  const float* wv = (const float*)d_in[8];
  const float* bv = (const float*)d_in[9];
  const float* wo = (const float*)d_in[10];
  const float* bo = (const float*)d_in[11];

  float* out  = (float*)d_out;                                  // [B,S,DM]
  float* attn = out + (size_t)BATCH * S_LEN * DM;               // [B,H,S,S]

  const size_t elems = (size_t)BATCH * NH * S_LEN * DEPTH;      // 8388608
  f16* ws  = (f16*)d_ws;
  f16* Qh  = ws;                 // [B,H,S,64]
  f16* Khb = ws + elems;         // [B,H,S,64]
  f16* Vt  = ws + 2 * elems;     // [B,H,64,S]
  f16* Ctx = ws + 3 * elems;     // [B,H,S,64]

  dim3 gg(BATCH * S_LEN / 128, DM / 128, 1);
  dim3 bb(256, 1, 1);
  gemm_kernel<0><<<gg, bb, 0, stream>>>(q, nullptr, wq, bq, Qh,  nullptr);
  gemm_kernel<0><<<gg, bb, 0, stream>>>(k, nullptr, wk, bk, Khb, nullptr);
  gemm_kernel<1><<<gg, bb, 0, stream>>>(v, nullptr, wv, bv, Vt,  nullptr);

  const size_t smem = (16 * 2064 + 16 * DEPTH) * sizeof(float);  // ~133 KB < 320 KB LDS
  attn_kernel<<<dim3(S_LEN / 16, NH, BATCH), bb, smem, stream>>>(Qh, Khb, Vt, attn, Ctx);

  gemm_kernel<2><<<gg, bb, 0, stream>>>(nullptr, Ctx, wo, bo, nullptr, out);
}
